// Transformer_71236327571528
// MI455X (gfx1250) — matrix operand order repaired and checked by reading, not compiled, not run on hardware
//
#include <hip/hip_runtime.h>

// ---------------- problem constants (fixed by the reference) ----------------
#define Dm   1024
#define NHm  16
#define HDm  64
#define FFm  4096
#define Lm   6
#define Vm   32000
#define NCm  10
#define Bm   4
#define Sm   1024
#define Tm   (Sm + 1)           // 1025 (CLS prepended)
#define BTm  (Bm * Tm)          // 4100 rows

typedef __attribute__((ext_vector_type(16))) __bf16 v16bf;
typedef __attribute__((ext_vector_type(8)))  __bf16 v8bf;
typedef __attribute__((ext_vector_type(8)))  float  v8f;

// fp32 -> bf16, round-to-nearest-even (bit level)
__device__ __forceinline__ __bf16 tobf(float f) {
    union { float f; unsigned u; } a; a.f = f;
    unsigned r = a.u + 0x7FFFu + ((a.u >> 16) & 1u);
    unsigned short h = (unsigned short)(r >> 16);
    __bf16 b; __builtin_memcpy(&b, &h, 2);
    return b;
}

// CDNA5 async copy: global -> LDS, 16 bytes per lane (tracked by ASYNCcnt)
__device__ __forceinline__ void async_load_b128(unsigned lds_off, const void* g) {
    unsigned long long ga = (unsigned long long)g;
    asm volatile("global_load_async_to_lds_b128 %0, %1, off"
                 :: "v"(lds_off), "v"(ga) : "memory");
}
__device__ __forceinline__ void wait_async0() {
    asm volatile("s_wait_asynccnt 0x0" ::: "memory");
}
// wait until only the 4 newest async ops remain (ASYNCcnt decrements in order)
__device__ __forceinline__ void wait_async4() {
    asm volatile("s_wait_asynccnt 0x4" ::: "memory");
}

// ---------------------------------------------------------------------------
// C[M,N] = A[M,K] @ W[N,K]^T (+bias, optional relu).  A,W bf16; out f32/bf16.
// Block = 8 waves, tile 128M x 128N, K-step 32.
// Double-buffered async LDS pipeline: issue slice k+1, wait slice k, compute.
// A-frag layout: lanes0-15 row=lane, elems0-15=K0..15;
//                lanes16-31 row=lane-16, elems0-15=K16..31.
// B-frag: lane=col(n), half kh: elems = K 16*kh..16*kh+15 (contiguous).
__global__ __launch_bounds__(256) void gemm_bf16(const __bf16* __restrict__ A,
                                                 const __bf16* __restrict__ W,
                                                 const float*  __restrict__ bias,
                                                 float*  __restrict__ Cf,     // nullable
                                                 __bf16* __restrict__ Cb,     // nullable
                                                 int M, int N, int K, int relu) {
    __shared__ __attribute__((aligned(32))) __bf16 smA[2][128 * 32];
    __shared__ __attribute__((aligned(32))) __bf16 smB[2][128 * 32];

    int tid  = threadIdx.x;
    int wave = tid >> 5;
    int lane = tid & 31;
    int m0 = blockIdx.x * 128;
    int n0 = blockIdx.y * 128;          // N is a multiple of 128
    int maxRow = M - 1;

    // staging: thread covers 8 bf16 at column aCol of rows aRow / aRow+64
    int aRow = tid >> 2;
    int aCol = (tid & 3) * 8;
    int ar0 = m0 + aRow;      ar0 = (ar0 > maxRow) ? maxRow : ar0;
    int ar1 = m0 + aRow + 64; ar1 = (ar1 > maxRow) ? maxRow : ar1;
    const __bf16* gA0 = A + (long)ar0 * K + aCol;
    const __bf16* gA1 = A + (long)ar1 * K + aCol;
    const __bf16* gB0 = W + (long)(n0 + aRow) * K + aCol;
    const __bf16* gB1 = W + (long)(n0 + aRow + 64) * K + aCol;

    unsigned offA0[2], offA1[2], offB0[2], offB1[2];
#pragma unroll
    for (int p = 0; p < 2; ++p) {
        offA0[p] = (unsigned)(uintptr_t)&smA[p][aRow * 32 + aCol];
        offA1[p] = (unsigned)(uintptr_t)&smA[p][(aRow + 64) * 32 + aCol];
        offB0[p] = (unsigned)(uintptr_t)&smB[p][aRow * 32 + aCol];
        offB1[p] = (unsigned)(uintptr_t)&smB[p][(aRow + 64) * 32 + aCol];
    }

    auto issue = [&](int p, int kk) {
        async_load_b128(offA0[p], gA0 + kk);
        async_load_b128(offA1[p], gA1 + kk);
        async_load_b128(offB0[p], gB0 + kk);
        async_load_b128(offB1[p], gB1 + kk);
    };

    int m = lane & 15, hf = lane >> 4;

    v8f acc[8] = {v8f{}, v8f{}, v8f{}, v8f{}, v8f{}, v8f{}, v8f{}, v8f{}};

    issue(0, 0);                         // prologue: slice 0 -> buffer 0
    int cur = 0;
    for (int k0 = 0; k0 < K; k0 += 32) {
        int nxt = cur ^ 1;
        int kn = k0 + 32; kn = (kn >= K) ? (K - 32) : kn;   // tail: redundant reload
        __builtin_prefetch(gB0 + kn + 32, 0, 0);            // global_prefetch_b8
        issue(nxt, kn);                  // stream slice k+1 behind the WMMAs
        wait_async4();                   // slice k complete (4 newest still flying)
        __syncthreads();

        const __bf16* Ac = smA[cur];
        const __bf16* Bc = smB[cur];
        v8bf alo = *(const v8bf*)&Ac[(wave * 16 + m) * 32 + 16 * hf];
        v8bf ahi = *(const v8bf*)&Ac[(wave * 16 + m) * 32 + 8 + 16 * hf];
        v16bf a = __builtin_shufflevector(alo, ahi, 0, 1, 2, 3, 4, 5, 6, 7,
                                          8, 9, 10, 11, 12, 13, 14, 15);
#pragma unroll
        for (int t = 0; t < 8; ++t) {
            v16bf b = *(const v16bf*)&Bc[(t * 16 + m) * 32 + 16 * hf];
            acc[t] = __builtin_amdgcn_wmma_f32_16x16x32_bf16(
                false, a, false, b, (short)0, acc[t], false, false);
        }
        __syncthreads();                 // all waves done reading 'cur' before reuse
        cur = nxt;
    }

    int col = lane & 15, rh = lane >> 4;
#pragma unroll
    for (int t = 0; t < 8; ++t) {
        int n = n0 + 16 * t + col;
        float bv = bias ? bias[n] : 0.0f;
#pragma unroll
        for (int r = 0; r < 8; ++r) {
            int row = m0 + wave * 16 + r + 8 * rh;
            if (row < M) {
                float v = acc[t][r] + bv;
                if (relu) v = fmaxf(v, 0.0f);
                if (Cf) Cf[(long)row * N + n] = v;
                if (Cb) Cb[(long)row * N + n] = tobf(v);
            }
        }
    }
}

// ---------------------------------------------------------------------------
// bulk fp32 -> bf16 (weights); n multiple of 4
__global__ __launch_bounds__(256) void cvt_bf16(const float* __restrict__ s,
                                                __bf16* __restrict__ d, long n) {
    long i = ((long)blockIdx.x * 256 + threadIdx.x) * 4;
    if (i < n) {
        float4 v = *(const float4*)(s + i);
        d[i]     = tobf(v.x);
        d[i + 1] = tobf(v.y);
        d[i + 2] = tobf(v.z);
        d[i + 3] = tobf(v.w);
    }
}

// ---------------------------------------------------------------------------
// x[b,t,:] = (t==0 ? cls : emb[token]) + pe(t)*D^-0.5 ; also bf16 mirror.
__global__ void embed_pe(const int* __restrict__ tokens,
                         const float* __restrict__ emb,
                         const float* __restrict__ cls,
                         float* __restrict__ x, __bf16* __restrict__ xb) {
    long idx = (long)blockIdx.x * blockDim.x + threadIdx.x;
    if (idx >= (long)BTm * Dm) return;
    int d = (int)(idx % Dm);
    long bt = idx / Dm;
    int t = (int)(bt % Tm);
    int b = (int)(bt / Tm);
    float base;
    if (t == 0) {
        base = cls[d];
    } else {
        int tok = tokens[(long)b * Sm + (t - 1)];
        base = emb[(long)tok * Dm + d];
    }
    int pair = d >> 1;
    float denom = __powf(10000.0f, (float)(2 * pair) / (float)Dm);
    float ang = (float)t / denom;
    float pe = ((d & 1) ? __cosf(ang) : __sinf(ang)) * 0.03125f;   // 1024^-0.5
    float v = base + pe;
    x[idx] = v;
    xb[idx] = tobf(v);
}

// ---------------------------------------------------------------------------
// x[row,:] = LN(x[row,:] + y[row,:]); also write bf16 mirror.
__global__ __launch_bounds__(256) void add_ln(float* __restrict__ x,
                                              const float* __restrict__ y,
                                              __bf16* __restrict__ xb) {
    long row = blockIdx.x;
    float* xr = x + row * Dm;
    const float* yr = y + row * Dm;
    __bf16* xbr = xb + row * Dm;
    float vloc[4];
    float s = 0.0f, s2 = 0.0f;
#pragma unroll
    for (int i = 0; i < 4; ++i) {
        int d = threadIdx.x + i * 256;
        float v = xr[d] + yr[d];
        vloc[i] = v; s += v; s2 += v * v;
    }
#pragma unroll
    for (int o = 16; o > 0; o >>= 1) { s += __shfl_xor(s, o); s2 += __shfl_xor(s2, o); }
    __shared__ float sh[16];
    int wid = threadIdx.x >> 5;
    if ((threadIdx.x & 31) == 0) { sh[wid] = s; sh[8 + wid] = s2; }
    __syncthreads();
    float S = 0.0f, S2 = 0.0f;
#pragma unroll
    for (int w = 0; w < 8; ++w) { S += sh[w]; S2 += sh[8 + w]; }
    float mu  = S * (1.0f / Dm);
    float var = S2 * (1.0f / Dm) - mu * mu;
    float inv = rsqrtf(var + 1e-5f);
#pragma unroll
    for (int i = 0; i < 4; ++i) {
        int d = threadIdx.x + i * 256;
        float v = (vloc[i] - mu) * inv;
        xr[d] = v;
        xbr[d] = tobf(v);
    }
}

// ---------------------------------------------------------------------------
// Flash attention, one wave per (b, h, 16-query tile); qkv is bf16
// [q(D)|k(D)|v(D)] per row, head h at column h*64 of each block.
// V tile (32 keys x 64 dims) staged via async copies, overlapped with S+softmax.
__global__ __launch_bounds__(32) void attn_flash(const __bf16* __restrict__ qkv,
                                                 __bf16* __restrict__ attn_out,
                                                 const int* __restrict__ seq_lens) {
    const int QT = (Tm + 15) / 16;          // 65
    int qt = blockIdx.x % QT;
    int bh = blockIdx.x / QT;
    int h = bh % NHm;
    int b = bh / NHm;
    int q0 = qt * 16;

    int lane = threadIdx.x & 31;
    int m = lane & 15, hf = lane >> 4;
    int col = lane & 15;

    int kvLimit = seq_lens[b] + 1;          // keys j < kvLimit attended (<= Tm)
    const float scale = 0.125f;             // 64^-0.5
    const long rs = 3 * Dm;
    const __bf16* Qb = qkv + (long)b * Tm * rs + h * HDm;
    const __bf16* Kb = Qb + Dm;
    const __bf16* Vb = Qb + 2 * Dm;

    // Q A-fragments (dims 0..31 and 32..63)
    v16bf qa0, qa1;
    {
        int r = q0 + m; r = (r >= Tm) ? Tm - 1 : r;
        const __bf16* p = Qb + (long)r * rs;
        v8bf l0 = *(const v8bf*)&p[16 * hf];
        v8bf h0 = *(const v8bf*)&p[8 + 16 * hf];
        v8bf l1 = *(const v8bf*)&p[32 + 16 * hf];
        v8bf h1 = *(const v8bf*)&p[40 + 16 * hf];
        qa0 = __builtin_shufflevector(l0, h0, 0,1,2,3,4,5,6,7,8,9,10,11,12,13,14,15);
        qa1 = __builtin_shufflevector(l1, h1, 0,1,2,3,4,5,6,7,8,9,10,11,12,13,14,15);
    }

    v8f o[4] = {v8f{}, v8f{}, v8f{}, v8f{}};
    float m_st[8], l_st[8];
#pragma unroll
    for (int r = 0; r < 8; ++r) { m_st[r] = -INFINITY; l_st[r] = 0.0f; }

    __shared__ __bf16 Pl[16 * 32];                      // P staging (D -> A layout)
    __shared__ __attribute__((aligned(32))) __bf16 Vt[32 * 64];   // V tile [key][dim]

    // async V staging addresses: lane covers dims (lane&7)*8, key row lane>>3
    int vDim = (lane & 7) * 8;
    int vRow = lane >> 3;

    for (int j0 = 0; j0 < kvLimit; j0 += 32) {
        // ---- kick off V tile copy (overlaps with S and softmax) ----
#pragma unroll
        for (int q = 0; q < 8; ++q) {
            int key = j0 + q * 4 + vRow; key = (key >= Tm) ? Tm - 1 : key;
            async_load_b128((unsigned)(uintptr_t)&Vt[(q * 4 + vRow) * 64 + vDim],
                            Vb + (long)key * rs + vDim);
        }

        // ---- S = Q K^T over 32 keys ----
        v8f s0 = v8f{}, s1 = v8f{};
        int keyA = j0 + m;      keyA = (keyA >= Tm) ? Tm - 1 : keyA;
        int keyB = j0 + 16 + m; keyB = (keyB >= Tm) ? Tm - 1 : keyB;
#pragma unroll
        for (int ks = 0; ks < 2; ++ks) {
            int kd = ks * 32;
            v16bf qa = ks ? qa1 : qa0;
            v16bf kb0 = *(const v16bf*)&Kb[(long)keyA * rs + kd + 16 * hf];
            s0 = __builtin_amdgcn_wmma_f32_16x16x32_bf16(
                false, qa, false, kb0, (short)0, s0, false, false);
            v16bf kb1 = *(const v16bf*)&Kb[(long)keyB * rs + kd + 16 * hf];
            s1 = __builtin_amdgcn_wmma_f32_16x16x32_bf16(
                false, qa, false, kb1, (short)0, s1, false, false);
        }

        // ---- mask + online softmax ----
        int ja = j0 + col, jb = j0 + 16 + col;
        float corr[8];
#pragma unroll
        for (int r = 0; r < 8; ++r) {
            float v0 = s0[r] * scale; if (ja >= kvLimit) v0 = -INFINITY;
            float v1 = s1[r] * scale; if (jb >= kvLimit) v1 = -INFINITY;
            float mx = fmaxf(v0, v1);
#pragma unroll
            for (int off = 8; off > 0; off >>= 1) mx = fmaxf(mx, __shfl_xor(mx, off));
            float nm = fmaxf(m_st[r], mx);
            corr[r] = __expf(m_st[r] - nm);
            m_st[r] = nm;
            float p0 = __expf(v0 - nm);
            float p1 = __expf(v1 - nm);
            float rsum = p0 + p1;
#pragma unroll
            for (int off = 8; off > 0; off >>= 1) rsum += __shfl_xor(rsum, off);
            l_st[r] = l_st[r] * corr[r] + rsum;
            int rowl = r + 8 * hf;
            Pl[rowl * 32 + col]      = tobf(p0);
            Pl[rowl * 32 + 16 + col] = tobf(p1);
        }
#pragma unroll
        for (int t = 0; t < 4; ++t)
#pragma unroll
            for (int r = 0; r < 8; ++r) o[t][r] *= corr[r];

        __syncthreads();
        v16bf pa;
        {
            v8bf plo = *(const v8bf*)&Pl[m * 32 + 16 * hf];
            v8bf phi = *(const v8bf*)&Pl[m * 32 + 8 + 16 * hf];
            pa = __builtin_shufflevector(plo, phi, 0,1,2,3,4,5,6,7,
                                         8,9,10,11,12,13,14,15);
        }
        __syncthreads();

        // ---- O += P V (V fragments gathered from LDS tile) ----
        wait_async0();
#pragma unroll
        for (int t = 0; t < 4; ++t) {
            v16bf vb;
#pragma unroll
            for (int i = 0; i < 8; ++i) {
                int k0e = 16 * hf + 2 * i;          // local key in tile
                vb[2 * i]     = Vt[k0e * 64 + t * 16 + m];
                vb[2 * i + 1] = Vt[(k0e + 1) * 64 + t * 16 + m];
            }
            o[t] = __builtin_amdgcn_wmma_f32_16x16x32_bf16(
                false, pa, false, vb, (short)0, o[t], false, false);
        }
    }

    // ---- normalize + store (bf16) ----
#pragma unroll
    for (int r = 0; r < 8; ++r) {
        int rowl = r + 8 * hf;
        int row = q0 + rowl;
        if (row < Tm) {
            float inv = 1.0f / l_st[r];
#pragma unroll
            for (int t = 0; t < 4; ++t) {
                attn_out[((long)b * Tm + row) * Dm + h * HDm + t * 16 + col] =
                    tobf(o[t][r] * inv);
            }
        }
    }
}

// ---------------------------------------------------------------------------
__global__ __launch_bounds__(32) void classifier_k(const float* __restrict__ x,
                                                   const float* __restrict__ Wc,
                                                   const float* __restrict__ bc,
                                                   float* __restrict__ out) {
    int b = blockIdx.x, c = blockIdx.y;
    const float* xr = x + (long)b * Tm * Dm;
    const float* wr = Wc + (long)c * Dm;
    float s = 0.0f;
    for (int d = threadIdx.x; d < Dm; d += 32) s += xr[d] * wr[d];
#pragma unroll
    for (int o = 16; o > 0; o >>= 1) s += __shfl_xor(s, o);
    if (threadIdx.x == 0) out[b * NCm + c] = s + bc[c];
}

// ---------------------------------------------------------------------------
extern "C" void kernel_launch(void* const* d_in, const int* in_sizes, int n_in,
                              void* d_out, int out_size, void* d_ws, size_t ws_size,
                              hipStream_t stream) {
    (void)in_sizes; (void)n_in; (void)out_size; (void)ws_size;
    const int*   tokens   = (const int*)d_in[0];
    const int*   seq_lens = (const int*)d_in[1];
    const float* emb      = (const float*)d_in[2];
    const float* cls      = (const float*)d_in[3];
    const float* Wqkv     = (const float*)d_in[4];
    const float* Wo       = (const float*)d_in[5];
    const float* W1       = (const float*)d_in[6];
    const float* b1       = (const float*)d_in[7];
    const float* W2       = (const float*)d_in[8];
    const float* b2       = (const float*)d_in[9];
    const float* Wc       = (const float*)d_in[10];
    const float* bc       = (const float*)d_in[11];
    float* out = (float*)d_out;

    const long nWq = (long)Lm * 3 * Dm * Dm;     // 18.9M
    const long nWo = (long)Lm * Dm * Dm;         //  6.3M
    const long nW1 = (long)Lm * FFm * Dm;        // 25.2M
    const long nW2 = (long)Lm * Dm * FFm;        // 25.2M

    char* base = (char*)d_ws;
    __bf16* wq_bf = (__bf16*)base;  base += nWq * 2;
    __bf16* wo_bf = (__bf16*)base;  base += nWo * 2;
    __bf16* w1_bf = (__bf16*)base;  base += nW1 * 2;
    __bf16* w2_bf = (__bf16*)base;  base += nW2 * 2;
    float*  x     = (float*)base;   base += (long)BTm * Dm * 4;
    __bf16* xb    = (__bf16*)base;  base += (long)BTm * Dm * 2;
    __bf16* qkvb  = (__bf16*)base;  base += (long)BTm * 3 * Dm * 2;
    __bf16* attnb = (__bf16*)base;  base += (long)BTm * Dm * 2;
    float*  proj  = (float*)base;   base += (long)BTm * Dm * 4;
    __bf16* hb    = (__bf16*)base;  base += (long)BTm * FFm * 2;
    float*  fb    = (float*)base;   base += (long)BTm * Dm * 4;

    // 0) weights -> bf16 (once per launch, amortized across all GEMM tile reuse)
    cvt_bf16<<<(int)((nWq / 4 + 255) / 256), 256, 0, stream>>>(Wqkv, wq_bf, nWq);
    cvt_bf16<<<(int)((nWo / 4 + 255) / 256), 256, 0, stream>>>(Wo,   wo_bf, nWo);
    cvt_bf16<<<(int)((nW1 / 4 + 255) / 256), 256, 0, stream>>>(W1,   w1_bf, nW1);
    cvt_bf16<<<(int)((nW2 / 4 + 255) / 256), 256, 0, stream>>>(W2,   w2_bf, nW2);

    // 1) embedding + CLS + positional encoding
    {
        long total = (long)BTm * Dm;
        embed_pe<<<(int)((total + 255) / 256), 256, 0, stream>>>(tokens, emb, cls, x, xb);
    }

    const int mtiles = (BTm + 127) / 128;        // 33
    const int QT = (Tm + 15) / 16;               // 65

    for (int l = 0; l < Lm; ++l) {
        const __bf16* Wqkv_l = wq_bf + (long)l * 3 * Dm * Dm;
        const __bf16* Wo_l   = wo_bf + (long)l * Dm * Dm;
        const __bf16* W1_l   = w1_bf + (long)l * FFm * Dm;
        const __bf16* W2_l   = w2_bf + (long)l * Dm * FFm;
        const float*  b1_l   = b1 + (long)l * FFm;
        const float*  b2_l   = b2 + (long)l * Dm;

        // qkv(bf16) = xb @ Wqkv^T
        gemm_bf16<<<dim3(mtiles, (3 * Dm) / 128), 256, 0, stream>>>(
            xb, Wqkv_l, nullptr, nullptr, qkvb, BTm, 3 * Dm, Dm, 0);
        // flash attention -> attnb (bf16)
        attn_flash<<<Bm * NHm * QT, 32, 0, stream>>>(qkvb, attnb, seq_lens);
        // proj(f32) = attnb @ Wo^T
        gemm_bf16<<<dim3(mtiles, Dm / 128), 256, 0, stream>>>(
            attnb, Wo_l, nullptr, proj, nullptr, BTm, Dm, Dm, 0);
        // x = LN(x + proj), xb mirror
        add_ln<<<BTm, 256, 0, stream>>>(x, proj, xb);
        // hb(bf16) = relu(xb @ W1^T + b1)
        gemm_bf16<<<dim3(mtiles, FFm / 128), 256, 0, stream>>>(
            xb, W1_l, b1_l, nullptr, hb, BTm, FFm, Dm, 1);
        // fb(f32) = hb @ W2^T + b2
        gemm_bf16<<<dim3(mtiles, Dm / 128), 256, 0, stream>>>(
            hb, W2_l, b2_l, fb, nullptr, BTm, Dm, FFm, 0);
        // x = LN(x + fb), xb mirror
        add_ln<<<BTm, 256, 0, stream>>>(x, fb, xb);
    }

    classifier_k<<<dim3(Bm, NCm), 32, 0, stream>>>(x, Wc, bc, out);
}
